// RNNEncoder_17239998726623
// MI455X (gfx1250) — matrix operand order repaired and checked by reading, not compiled, not run on hardware
//
#include <hip/hip_runtime.h>

typedef __attribute__((ext_vector_type(16))) __bf16 v16bf;
typedef __attribute__((ext_vector_type(8)))  __bf16 v8bf;
typedef __attribute__((ext_vector_type(8)))  float  v8f;

#define B_    512
#define T_    512
#define H_    256
#define LDSTR 264                      // padded bf16 row stride of the h tile (528 B)
#define HTILE (16 * LDSTR)             // one h buffer, elements
#define WGATE (16 * 8 * 2 * 32 * 8)    // one gate of pre-swizzled W_hh fragments, elements (128 KB)

__device__ __forceinline__ float sigmoidf_(float x) {
    return 1.0f / (1.0f + __expf(-x));
}
__device__ __forceinline__ float tanhf_(float x) {
    float ax = fabsf(x);
    float e  = __expf(2.0f * ax);
    float t  = 1.0f - 2.0f / (e + 1.0f);
    return copysignf(t, x);
}

__device__ __forceinline__ v8f wmma_bf16(v16bf a, v16bf b, v8f c) {
    return __builtin_amdgcn_wmma_f32_16x16x32_bf16(false, a, false, b, (short)0, c, false, false);
}

__global__ __launch_bounds__(512, 1)
void gru_scan_wmma(const float* __restrict__ inputs,     // [B][T][2]
                   const int*   __restrict__ input_len,  // [B]
                   const float* __restrict__ h0,         // [B][H]
                   const float* __restrict__ W_ih,       // [3H][2]
                   const float* __restrict__ W_hh,       // [3H][H]
                   const float* __restrict__ b_ih,       // [3H]
                   const float* __restrict__ b_hh,       // [3H]
                   float*       __restrict__ out)        // [B][H]
{
    extern __shared__ __align__(16) __bf16 smem[];
    __bf16* hbuf = smem;                 // [2][16*LDSTR]   double-buffered h tile (bf16)
    __bf16* wsm  = smem + 2 * HTILE;     // [2 gates][16 w][8 kb][2 half][32 lane][8] W_hh z/n

    const int lane  = threadIdx.x & 31;
    const int wave  = threadIdx.x >> 5;     // 0..15, owns hidden cols [16*wave, 16*wave+16)
    const int nloc  = lane & 15;
    const int khalf = lane >> 4;            // 0 or 1
    const int j     = wave * 16 + nloc;     // this lane's hidden column (0..255)
    const int rowb  = blockIdx.x * 16;      // batch-row base of this workgroup

    // ---- r-gate W_hh fragments stay in VGPRs; z/n gates go to LDS (pre-swizzled) --------
    // B (K x N) layout: lane<16 -> N=lane, K in [k0,k0+16); lane>=16 -> N=lane-16, K += 16
    v16bf bfr[8];
#pragma unroll
    for (int kb = 0; kb < 8; ++kb) {
        const int k0 = kb * 32 + khalf * 16;
        const float* pr = W_hh + (         j) * H_ + k0;
        const float* pz = W_hh + (H_     + j) * H_ + k0;
        const float* pn = W_hh + (2 * H_ + j) * H_ + k0;
        v16bf z, n;
#pragma unroll
        for (int e = 0; e < 16; ++e) {
            bfr[kb][e] = (__bf16)pr[e];
            z[e]       = (__bf16)pz[e];
            n[e]       = (__bf16)pn[e];
        }
        // slot(gate,wave,kb,half,lane) = (((gate*16+wave)*8+kb)*2+half)*256 + lane*8 elems
        union { v16bf v; v8bf h[2]; } uz, un;
        uz.v = z; un.v = n;
        const int base = ((wave * 8 + kb) * 2) * 256 + lane * 8;
        *(v8bf*)(wsm + base        )          = uz.h[0];
        *(v8bf*)(wsm + base + 256  )          = uz.h[1];
        *(v8bf*)(wsm + WGATE + base)          = un.h[0];
        *(v8bf*)(wsm + WGATE + base + 256)    = un.h[1];
    }

    // ---- per-lane gate scalars (input projection + biases) -------------------------------
    const float wr0 = W_ih[(         j) * 2 + 0], wr1 = W_ih[(         j) * 2 + 1];
    const float wz0 = W_ih[(H_     + j) * 2 + 0], wz1 = W_ih[(H_     + j) * 2 + 1];
    const float wn0 = W_ih[(2 * H_ + j) * 2 + 0], wn1 = W_ih[(2 * H_ + j) * 2 + 1];
    const float br  = b_ih[         j] + b_hh[         j];
    const float bz  = b_ih[H_     + j] + b_hh[H_     + j];
    const float bni = b_ih[2 * H_ + j];     // n-gate biases stay split (r scales gh_n+b_hh_n)
    const float bnh = b_hh[2 * H_ + j];

    // ---- per-lane h-state fragment in WMMA C layout: VGPR g -> row g+8*khalf, col j ------
    float hfrag[8];
    int   len8[8];
    int   rowT[8];
    const int hwbase = (8 * khalf) * LDSTR + j;
#pragma unroll
    for (int g = 0; g < 8; ++g) {
        const int row = rowb + g + 8 * khalf;
        hfrag[g] = h0[row * H_ + j];
        len8[g]  = input_len[row];
        rowT[g]  = row * T_;
        hbuf[hwbase + g * LDSTR] = (__bf16)hfrag[g];        // seed buffer 0
    }
    __syncthreads();

    const float2* xbase = (const float2*)inputs;            // [B*T] float2 (IN==2)
    const __bf16* wzb = wsm         + (wave * 8 * 2) * 256 + lane * 8;
    const __bf16* wnb = wsm + WGATE + (wave * 8 * 2) * 256 + lane * 8;

    for (int t = 0; t < T_; ++t) {
        const int p = t & 1;
        const __bf16* hrd = hbuf + p * HTILE;
        __bf16*       hwr = hbuf + (p ^ 1) * HTILE;

        // hoist x loads: latency hidden under the 24-WMMA phase
        float2 xv[8];
#pragma unroll
        for (int g = 0; g < 8; ++g) xv[g] = xbase[rowT[g] + t];

        // -------- gh = h @ W_hh^T for this wave's 3x16 output columns ---------------------
        v8f accr = {}, accz = {}, accn = {};
#pragma unroll
        for (int kb = 0; kb < 8; ++kb) {
            // A-fragment (16x32 bf16): lane<16 row=lane K=32kb+{0..15};
            //                          lane>=16 row=lane-16 K=32kb+{16..31}
            union { v16bf v; v8bf h[2]; } ua, uz, un;
            const __bf16* ap = hrd + nloc * LDSTR + kb * 32 + khalf * 16;
            ua.h[0] = *(const v8bf*)(ap);
            ua.h[1] = *(const v8bf*)(ap + 8);
            uz.h[0] = *(const v8bf*)(wzb + kb * 512);
            uz.h[1] = *(const v8bf*)(wzb + kb * 512 + 256);
            un.h[0] = *(const v8bf*)(wnb + kb * 512);
            un.h[1] = *(const v8bf*)(wnb + kb * 512 + 256);
            accr = wmma_bf16(ua.v, bfr[kb], accr);
            accz = wmma_bf16(ua.v, uz.v,   accz);
            accn = wmma_bf16(ua.v, un.v,   accn);
        }

        // -------- gates + state update (f32) ----------------------------------------------
#pragma unroll
        for (int g = 0; g < 8; ++g) {
            const float2 x  = xv[g];
            const float gir = fmaf(x.y, wr1, fmaf(x.x, wr0, br));
            const float giz = fmaf(x.y, wz1, fmaf(x.x, wz0, bz));
            const float gin = fmaf(x.y, wn1, fmaf(x.x, wn0, bni));
            const float r  = sigmoidf_(accr[g] + gir);
            const float z  = sigmoidf_(accz[g] + giz);
            const float n  = tanhf_(gin + r * (accn[g] + bnh));
            const float hn = (1.0f - z) * n + z * hfrag[g];
            hfrag[g] = hn;
            hwr[hwbase + g * LDSTR] = (__bf16)hn;           // publish h(t+1) to other buffer
            if (t == len8[g] - 1) {                         // capture hs[len-1] exactly once
                out[(rowb + g + 8 * khalf) * H_ + j] = hn;
            }
        }
        __syncthreads();   // single barrier: h(t+1) visible; everyone done with hbuf[p]
    }
}

extern "C" void kernel_launch(void* const* d_in, const int* in_sizes, int n_in,
                              void* d_out, int out_size, void* d_ws, size_t ws_size,
                              hipStream_t stream) {
    (void)in_sizes; (void)n_in; (void)out_size; (void)d_ws; (void)ws_size;
    const float* inputs    = (const float*)d_in[0];
    const int*   input_len = (const int*)  d_in[1];
    const float* h0        = (const float*)d_in[2];
    const float* W_ih      = (const float*)d_in[3];
    const float* W_hh      = (const float*)d_in[4];
    const float* b_ih      = (const float*)d_in[5];
    const float* b_hh      = (const float*)d_in[6];
    float*       out       = (float*)d_out;

    const size_t lds_bytes = (2 * (size_t)HTILE + 2 * (size_t)WGATE) * sizeof(__bf16); // ~273 KB
    dim3 grid(B_ / 16);     // 32 workgroups, 16 batch rows each (fully independent scans)
    dim3 block(512);        // 16 waves of 32
    hipLaunchKernelGGL(gru_scan_wmma, grid, block, lds_bytes, stream,
                       inputs, input_len, h0, W_ih, W_hh, b_ih, b_hh, out);
}
